// DREAMCellOptimized_37572373905842
// MI455X (gfx1250) — compile-verified, matrix operand order checked by reading, not run on hardware
//
#include <hip/hip_runtime.h>
#include <hip/hip_bf16.h>
#include <math.h>
#include <stdint.h>

// ---------------------------------------------------------------------------
// DREAM recurrent scan for MI455X (gfx1250, wave32, WMMA, async-to-LDS).
// 32 persistent workgroups: 2 batch tiles (M=16) x 16 slice owners.
// Each WG owns: H-slice of 32 (U state, LTC update) and I-slice of 16 (x_pred).
// Weight slices LDS-resident (one-time global_load_async_to_lds_b128 DMA).
// Cross-WG exchange uses DEV-scoped accesses; barriers fence with
// s_wait_storecnt only (no WGP$ writeback/invalidate).
// ---------------------------------------------------------------------------

typedef __attribute__((ext_vector_type(16))) __bf16 v16bf;
typedef __attribute__((ext_vector_type(8)))  float  v8f;

#define HID   512
#define INP   256
#define RNK   64
#define TSTEPS 2048
#define NWG   16
#define HSL   32
#define ISL   16
#define NBLK  32
#define NTHR  256

#define BETA_C   0.1f
#define DT_C     0.1f
#define FORGET_C 0.1f
#define TGT_C    1.0f
#define KAPPA_C  0.5f
#define ALPHA_C  0.1f
#define GAMMA_C  1.0f
#define TAUSYS_C 1.0f
#define TAUSS_C  1.0f
#define EPS_C    1e-6f

// workspace byte offsets
#define OFF_C     0
#define OFF_W     262144
#define OFF_B     524288
#define OFF_V     786432
#define OFF_VSUM  819200
#define OFF_BAR   819456
#define OFF_HX    819712
#define OFF_SUMSQ 885248
#define OFF_ERRX  887296
#define OFF_U     920576

__device__ __forceinline__ int kmap(int ln, int j) {
    return (j & 7) + (((j >> 3) & 1) << 4) + (((ln >> 4) & 1) << 3);
}

__device__ __forceinline__ float fast_tanh(float x) {
#if __has_builtin(__builtin_amdgcn_tanhf)
    return __builtin_amdgcn_tanhf(x);      // native v_tanh_f32 on gfx1250
#else
    return tanhf(x);
#endif
}

__device__ __forceinline__ void wait_async0() {
#if __has_builtin(__builtin_amdgcn_s_wait_asynccnt)
    __builtin_amdgcn_s_wait_asynccnt(0);
#else
    asm volatile("s_wait_asynccnt 0" ::: "memory");
#endif
}

__device__ __forceinline__ void async_g2l_b128(uint32_t lds_addr, const char* gaddr) {
    asm volatile("global_load_async_to_lds_b128 %0, %1, off"
                 :: "v"(lds_addr), "v"(gaddr) : "memory");
}

// DEV-scoped exchange accesses (bypass WGP$, land in / read from global L2)
__device__ __forceinline__ void st_dev(float* p, float v) {
    __hip_atomic_store(p, v, __ATOMIC_RELAXED, __HIP_MEMORY_SCOPE_AGENT);
}
__device__ __forceinline__ float ld_dev(const float* p) {
    return __hip_atomic_load(p, __ATOMIC_RELAXED, __HIP_MEMORY_SCOPE_AGENT);
}
// stores ack'd at their scope point -> sufficient pre-barrier fence
__device__ __forceinline__ void fence_store_dev() {
    asm volatile("s_wait_storecnt 0x0" ::: "memory");
}

// ---------------------------------------------------------------------------
// Prelude: f32 weights -> bf16 WMMA-fragment-swizzled global layouts + vsum
// ---------------------------------------------------------------------------
__global__ void dream_prep(const float* __restrict__ C, const float* __restrict__ W,
                           const float* __restrict__ Bb, const float* __restrict__ V,
                           char* __restrict__ ws) {
    if (blockIdx.x == 1600) {
        int t = threadIdx.x;
        float* vs = (float*)(ws + OFF_VSUM);
        if (t < RNK) {
            float s = 0.0f;
            for (int k = 0; k < INP; ++k) s += V[k * RNK + t];
            vs[t] = s;
        }
        if (t == 64 || t == 65) ((int*)(ws + OFF_BAR))[t - 64] = 0;
        return;
    }
    long f = (long)blockIdx.x * NTHR + threadIdx.x;
    const float* src; __bf16* dst; int NC, NSRC;
    if (f < 131072)      { src = C;  dst = (__bf16*)(ws + OFF_C); NC = 16; NSRC = INP; }
    else if (f < 262144) { f -= 131072; src = W;  dst = (__bf16*)(ws + OFF_W); NC = 32; NSRC = HID; }
    else if (f < 393216) { f -= 262144; src = Bb; dst = (__bf16*)(ws + OFF_B); NC = 32; NSRC = HID; }
    else                 { f -= 393216; src = V;  dst = (__bf16*)(ws + OFF_V); NC = 4;  NSRC = RNK; }
    int tc = (int)(f >> 9), r = (int)(f & 511);
    int ln = r >> 4, j = r & 15;
    int kc = tc / NC, nc = tc % NC;
    int k = kc * 32 + kmap(ln, j);
    int n = nc * 16 + (ln & 15);
    dst[f] = (__bf16)src[k * NSRC + n];
}

// ---------------------------------------------------------------------------
// Global sense-reversing barrier across 32 persistent workgroups (L2 atomics)
// ---------------------------------------------------------------------------
__device__ __forceinline__ void grid_barrier(int* bar) {
    fence_store_dev();
    __syncthreads();
    if (threadIdx.x == 0) {
        int g = __hip_atomic_load(&bar[1], __ATOMIC_RELAXED, __HIP_MEMORY_SCOPE_AGENT);
        int a = __hip_atomic_fetch_add(&bar[0], 1, __ATOMIC_ACQ_REL, __HIP_MEMORY_SCOPE_AGENT);
        if (a == NBLK - 1) {
            __hip_atomic_store(&bar[0], 0, __ATOMIC_RELAXED, __HIP_MEMORY_SCOPE_AGENT);
            __hip_atomic_fetch_add(&bar[1], 1, __ATOMIC_RELEASE, __HIP_MEMORY_SCOPE_AGENT);
        } else {
            while (__hip_atomic_load(&bar[1], __ATOMIC_ACQUIRE, __HIP_MEMORY_SCOPE_AGENT) == g) {
                __builtin_amdgcn_s_sleep(1);
            }
        }
    }
    __syncthreads();
}

// ---------------------------------------------------------------------------
// Main persistent scan kernel
// ---------------------------------------------------------------------------
__global__ __launch_bounds__(NTHR)
void dream_main(const float* __restrict__ x_seq, const float* __restrict__ eta,
                char* __restrict__ ws, float* __restrict__ out) {
    const int tile = blockIdx.x >> 4;
    const int wg   = blockIdx.x & 15;
    const int tid  = threadIdx.x;
    const int lane = tid & 31;
    const int wave = tid >> 5;

    const float* vsg = (const float*)(ws + OFF_VSUM);
    int*   bar   = (int*)(ws + OFF_BAR);
    float* h_x   = (float*)(ws + OFF_HX);
    float* ssq_x = (float*)(ws + OFF_SUMSQ);
    float* err_x = (float*)(ws + OFF_ERRX);
    float* U     = (float*)(ws + OFF_U);

    __shared__ __align__(32) __bf16 sh_Cs[16 * 512];
    __shared__ __align__(32) __bf16 sh_Ws[16 * 512];
    __shared__ __align__(32) __bf16 sh_Bs[16 * 512];
    __shared__ __align__(32) __bf16 sh_Vs[32 * 512];
    __shared__ __align__(32) __bf16 sh_hA[16 * 512];
    __shared__ __align__(32) __bf16 sh_xA[8 * 512];
    __shared__ __align__(32) __bf16 sh_eA[8 * 512];
    __shared__ __align__(16) float  sh_part[8 * 256];
    __shared__ float sh_eV[16 * RNK];
    __shared__ float sh_hown[16 * HSL];
    __shared__ float sh_fast[16 * HSL];
    __shared__ float sh_base[16 * HSL];
    __shared__ float sh_errW[16 * HSL];
    __shared__ float sh_red[16 * 3];
    __shared__ float sh_sur[16], sh_gain[16], sh_dtt[16], sh_scale[16], sh_pscale[16];
    __shared__ float sh_vsum[RNK];

    const int b_my = tid >> 4;
    const int hh0  = (tid & 15) * 2;
    const float eta0 = eta[wg * HSL + hh0];
    const float eta1 = eta[wg * HSL + hh0 + 1];

    float em[16], ev[16];
#pragma unroll
    for (int j = 0; j < 16; ++j) { em[j] = 0.0f; ev[j] = 1.0f; }

    // ---- one-time: async-DMA weight slices from global into LDS ------------
    {
        const uint32_t lcs = (uint32_t)(uintptr_t)sh_Cs;
        const uint32_t lws = (uint32_t)(uintptr_t)sh_Ws;
        const uint32_t lbs = (uint32_t)(uintptr_t)sh_Bs;
        const uint32_t lvs = (uint32_t)(uintptr_t)sh_Vs;
        for (int u = tid; u < 5120; u += NTHR) {
            uint32_t la; uint64_t go;
            if (u < 1024) {
                int idx = u, tl = idx >> 6, rb = (idx & 63) * 16;
                la = lcs + (uint32_t)idx * 16;
                go = (uint64_t)OFF_C + (uint64_t)(tl * 16 + wg) * 1024u + rb;
            } else if (u < 2048) {
                int idx = u - 1024, tl = idx >> 6, rb = (idx & 63) * 16;
                la = lws + (uint32_t)idx * 16;
                go = (uint64_t)OFF_W + (uint64_t)((tl >> 1) * 32 + wg * 2 + (tl & 1)) * 1024u + rb;
            } else if (u < 3072) {
                int idx = u - 2048, tl = idx >> 6, rb = (idx & 63) * 16;
                la = lbs + (uint32_t)idx * 16;
                go = (uint64_t)OFF_B + (uint64_t)((tl >> 1) * 32 + wg * 2 + (tl & 1)) * 1024u + rb;
            } else {
                int idx = u - 3072;
                la = lvs + (uint32_t)idx * 16;
                go = (uint64_t)OFF_V + (uint64_t)idx * 16;
            }
            async_g2l_b128(la, ws + go);
        }
        wait_async0();
    }

    // init LDS state + own U chunk
    for (int f = tid; f < 16 * 512; f += NTHR) sh_hA[f] = (__bf16)0.0f;
    for (int f = tid; f < 16 * HSL; f += NTHR) sh_hown[f] = 0.0f;
    if (tid < 16) sh_pscale[tid] = 1.0f;
    if (tid < RNK) sh_vsum[tid] = vsg[tid];
    for (int f = tid; f < 16 * HSL * RNK; f += NTHR) {
        int b = f / (HSL * RNK);
        int rem = f - b * (HSL * RNK);
        int hh = rem / RNK, r = rem - hh * RNK;
        U[(((size_t)(tile * 16 + b)) * HID + wg * HSL + hh) * RNK + r] = 0.0f;
    }
    // prologue: x A-fragments for ts = 0
    for (int f = tid; f < 8 * 512; f += NTHR) {
        int c = f >> 9, r2 = f & 511, ln = r2 >> 4, j = r2 & 15;
        int m = ln & 15;
        int i = c * 32 + kmap(ln, j);
        sh_xA[f] = (__bf16)x_seq[((size_t)(tile * 16 + m) * TSTEPS + 0) * INP + i];
    }
    __syncthreads();

    for (int ts = 0; ts < TSTEPS; ++ts) {
        // ---- Phase 1: x_pred I-slice, kc split over all 8 waves ----------
        {
            v8f acc = {0.f, 0.f, 0.f, 0.f, 0.f, 0.f, 0.f, 0.f};
#pragma unroll
            for (int kk = 0; kk < 2; ++kk) {
                int kc = wave * 2 + kk;
                v16bf a = *(const v16bf*)(sh_hA + (kc * 32 + lane) * 16);
                v16bf b = *(const v16bf*)(sh_Cs + (kc * 32 + lane) * 16);
                acc = __builtin_amdgcn_wmma_f32_16x16x32_bf16(
                          false, a, false, b, (short)0, acc, false, false);
            }
            int n = lane & 15, mb = (lane >> 4) << 3;
#pragma unroll
            for (int g = 0; g < 8; ++g) sh_part[(wave * 16 + mb + g) * 16 + n] = acc[g];
        }
        __syncthreads();
        {   // reduce partials, error slice -> DEV-scoped exchange
            int m = tid >> 4, n = tid & 15;
            float s = 0.0f;
#pragma unroll
            for (int w = 0; w < 8; ++w) s += sh_part[(w * 16 + m) * 16 + n];
            float xp = fast_tanh(s);
            int i = wg * ISL + n;
            float xv = x_seq[((size_t)(tile * 16 + m) * TSTEPS + ts) * INP + i];
            st_dev(&err_x[(tile * 16 + m) * INP + i], xv - xp);
        }
        grid_barrier(bar);

        // ---- Phase 2: full error frags + per-batch stats -----------------
        if (tid < 48) sh_red[tid] = 0.0f;
        for (int f = tid; f < 8 * 512; f += NTHR) {
            int c = f >> 9, r2 = f & 511, ln = r2 >> 4, j = r2 & 15;
            int m = ln & 15;
            int i = c * 32 + kmap(ln, j);
            sh_eA[f] = (__bf16)ld_dev(&err_x[(tile * 16 + m) * INP + i]);
        }
        __syncthreads();
        {
            float pe2 = 0.f, pm2 = 0.f, pev = 0.f;
            int ib = (tid & 15) * 16;
            const float* er = err_x + (tile * 16 + b_my) * INP + ib;
#pragma unroll
            for (int j = 0; j < 16; ++j) {
                float e = ld_dev(&er[j]);
                pe2 += e * e;
                pm2 += em[j] * em[j];
                pev += ev[j];
                float emn = (1.0f - BETA_C) * em[j] + BETA_C * e;
                float d = e - emn;
                ev[j] = (1.0f - BETA_C) * ev[j] + BETA_C * d * d;
                em[j] = emn;
            }
            atomicAdd(&sh_red[b_my * 3 + 0], pe2);
            atomicAdd(&sh_red[b_my * 3 + 1], pm2);
            atomicAdd(&sh_red[b_my * 3 + 2], pev);
        }
        __syncthreads();
        if (tid < 16) {
            int b = tid;
            float en  = sqrtf(sh_red[b * 3 + 0]);
            float bl  = sqrtf(sh_red[b * 3 + 1]) + EPS_C;
            float var = sh_red[b * 3 + 2] * (1.0f / INP);
            float ent = 0.5f * __logf(2.0f * 3.14159265358979f * (var + EPS_C));
            ent = fminf(fmaxf(ent, 0.0f), 2.0f);
            float tau = 1.0f + ALPHA_C * ent;
            float z   = (en / bl - tau) / (GAMMA_C * 2.0f);
            float sur = 1.0f / (1.0f + __expf(-z));
            sh_sur[b]  = sur;
            sh_gain[b] = 1.0f + KAPPA_C * sur;
            float tl = fminf(fmaxf(TAUSYS_C / (1.0f + sur * TAUSS_C), 0.01f), 50.0f);
            sh_dtt[b] = fminf(fmaxf(DT_C / (tl + DT_C), 0.01f), 0.5f);
        }
        __syncthreads();

        // ---- Phase 3: eV = err@V (waves 0-3), base (4-5), errW (6-7) -----
        {
            v8f acc = {0.f, 0.f, 0.f, 0.f, 0.f, 0.f, 0.f, 0.f};
            int mb = (lane >> 4) << 3;
            if (wave < 4) {
                int nc = wave;
#pragma unroll 4
                for (int kc = 0; kc < 8; ++kc) {
                    v16bf a = *(const v16bf*)(sh_eA + (kc * 32 + lane) * 16);
                    v16bf b = *(const v16bf*)(sh_Vs + ((kc * 4 + nc) * 32 + lane) * 16);
                    acc = __builtin_amdgcn_wmma_f32_16x16x32_bf16(
                              false, a, false, b, (short)0, acc, false, false);
                }
                int n = nc * 16 + (lane & 15);
#pragma unroll
                for (int g = 0; g < 8; ++g) sh_eV[(mb + g) * RNK + n] = acc[g];
            } else if (wave < 6) {
                int l = wave - 4;
#pragma unroll 4
                for (int kc = 0; kc < 8; ++kc) {
                    v16bf a = *(const v16bf*)(sh_xA + (kc * 32 + lane) * 16);
                    v16bf b = *(const v16bf*)(sh_Bs + ((kc * 2 + l) * 32 + lane) * 16);
                    acc = __builtin_amdgcn_wmma_f32_16x16x32_bf16(
                              false, a, false, b, (short)0, acc, false, false);
                }
                int n = l * 16 + (lane & 15);
#pragma unroll
                for (int g = 0; g < 8; ++g) sh_base[(mb + g) * HSL + n] = acc[g];
            } else {
                int l = wave - 6;
#pragma unroll 4
                for (int kc = 0; kc < 8; ++kc) {
                    v16bf a = *(const v16bf*)(sh_eA + (kc * 32 + lane) * 16);
                    v16bf b = *(const v16bf*)(sh_Ws + ((kc * 2 + l) * 32 + lane) * 16);
                    acc = __builtin_amdgcn_wmma_f32_16x16x32_bf16(
                              false, a, false, b, (short)0, acc, false, false);
                }
                int n = l * 16 + (lane & 15);
#pragma unroll
                for (int g = 0; g < 8; ++g) sh_errW[(mb + g) * HSL + n] = acc[g];
            }
        }
        __syncthreads();

        // ---- Phase 4: U update + pipelined x frags for ts+1 --------------
        if (tid < 16) sh_red[tid] = 0.0f;
        __syncthreads();
        {
            float ps  = sh_pscale[b_my];
            float sur = sh_sur[b_my];
            float ss  = 0.0f;
#pragma unroll 1
            for (int q = 0; q < 2; ++q) {
                int hh = hh0 + q;
                float hcoef = (q ? eta1 : eta0) * sur * sh_hown[b_my * HSL + hh];
                float fsum = 0.0f;
                float* up = U + (((size_t)(tile * 16 + b_my)) * HID + wg * HSL + hh) * RNK;
#pragma unroll 4
                for (int r = 0; r < RNK; r += 4) {
                    float4 u4 = *(const float4*)(up + r);
                    float uu[4] = {u4.x, u4.y, u4.z, u4.w};
#pragma unroll
                    for (int c = 0; c < 4; ++c) {
                        float u  = uu[c] * ps;
                        float un = u + DT_C * (hcoef * sh_eV[b_my * RNK + r + c] - FORGET_C * u);
                        uu[c] = un;
                        ss   += un * un;
                        fsum += un * sh_vsum[r + c];
                    }
                    u4.x = uu[0]; u4.y = uu[1]; u4.z = uu[2]; u4.w = uu[3];
                    *(float4*)(up + r) = u4;
                }
                sh_fast[b_my * HSL + hh] = fsum;
            }
            atomicAdd(&sh_red[b_my], ss);
        }
        if (ts + 1 < TSTEPS) {   // hide x gather latency under sumsq barrier
            for (int f = tid; f < 8 * 512; f += NTHR) {
                int c = f >> 9, r2 = f & 511, ln = r2 >> 4, j = r2 & 15;
                int m = ln & 15;
                int i = c * 32 + kmap(ln, j);
                sh_xA[f] = (__bf16)x_seq[((size_t)(tile * 16 + m) * TSTEPS + ts + 1) * INP + i];
            }
        }
        __syncthreads();
        if (tid < 16) st_dev(&ssq_x[(tile * NWG + wg) * 16 + tid], sh_red[tid]);
        grid_barrier(bar);

        // ---- Phase 5: reduce U norm -> scale -----------------------------
        if (tid < 16) {
            float s = 0.0f;
            for (int w = 0; w < NWG; ++w) s += ld_dev(&ssq_x[(tile * NWG + w) * 16 + tid]);
            float sc = fminf(TGT_C / (sqrtf(s) + EPS_C), 2.0f);
            sh_scale[tid]  = sc;
            sh_pscale[tid] = sc;
        }
        __syncthreads();

        // ---- Phase 6: LTC hidden update for owned slice ------------------
        {
            float dtt = sh_dtt[b_my], gn = sh_gain[b_my], sc = sh_scale[b_my];
#pragma unroll
            for (int q = 0; q < 2; ++q) {
                int hh = hh0 + q;
                float ue = gn * sh_base[b_my * HSL + hh] + 0.1f * sc * sh_fast[b_my * HSL + hh];
                float hp = sh_hown[b_my * HSL + hh];
                float hl = (1.0f - dtt) * hp + dtt * fast_tanh(ue);
                float hn = (hl + sh_errW[b_my * HSL + hh]) * 0.99f + hp * 0.01f;
                st_dev(&h_x[(tile * 16 + b_my) * HID + wg * HSL + hh], hn);
                __builtin_nontemporal_store(
                    hn, &out[((size_t)(tile * 16 + b_my) * TSTEPS + ts) * HID + wg * HSL + hh]);
            }
        }
        grid_barrier(bar);

        // ---- Phase 7: refresh full h fragments ---------------------------
        for (int f = tid; f < 16 * HID; f += NTHR) {
            int b = f >> 9, hg = f & 511;
            float v = ld_dev(&h_x[(tile * 16 + b) * HID + hg]);
            int c = hg >> 5, K = hg & 31;
            int ln2 = b + (((K >> 3) & 1) << 4);
            int j2  = (K & 7) + (((K >> 4) & 1) << 3);
            sh_hA[(c * 32 + ln2) * 16 + j2] = (__bf16)v;
            int hl = hg - wg * HSL;
            if ((unsigned)hl < HSL) sh_hown[b * HSL + hl] = v;
        }
        __syncthreads();
    }
}

extern "C" void kernel_launch(void* const* d_in, const int* in_sizes, int n_in,
                              void* d_out, int out_size, void* d_ws, size_t ws_size,
                              hipStream_t stream) {
    const float* x_seq = (const float*)d_in[0];
    const float* C     = (const float*)d_in[1];
    const float* W     = (const float*)d_in[2];
    const float* Bb    = (const float*)d_in[3];
    const float* V     = (const float*)d_in[4];
    const float* eta   = (const float*)d_in[5];
    char* ws = (char*)d_ws;

    dream_prep<<<1601, NTHR, 0, stream>>>(C, W, Bb, V, ws);
    dream_main<<<NBLK, NTHR, 0, stream>>>(x_seq, eta, ws, (float*)d_out);
}
